// MinkowskiRCNNSpMiddleFHD_7086696038821
// MI455X (gfx1250) — matrix operand order repaired and checked by reading, not compiled, not run on hardware
//
#include <hip/hip_runtime.h>
#include <stdint.h>

// ---------- types ----------
typedef __attribute__((ext_vector_type(16))) __bf16 v16bf;
typedef __attribute__((ext_vector_type(8)))  float  v8f;

__device__ __forceinline__ float bf2f(unsigned short h) {
  unsigned int u = ((unsigned int)h) << 16;
  float f; __builtin_memcpy(&f, &u, 4); return f;
}
__device__ __forceinline__ unsigned short f2bf(float f) {
  unsigned int u; __builtin_memcpy(&u, &f, 4);
  u += 0x7FFFu + ((u >> 16) & 1u);          // round-to-nearest-even
  return (unsigned short)(u >> 16);
}

// ---------- utility kernels ----------
__global__ void zero_u4(uint4* p, long long n) {
  uint4 z = make_uint4(0u, 0u, 0u, 0u);
  for (long long i = blockIdx.x * (long long)blockDim.x + threadIdx.x; i < n;
       i += (long long)gridDim.x * blockDim.x)
    p[i] = z;
}
__global__ void zero_f32(float* p, long long n) {
  for (long long i = blockIdx.x * (long long)blockDim.x + threadIdx.x; i < n;
       i += (long long)gridDim.x * blockDim.x)
    p[i] = 0.0f;
}

// scatter fp32 features -> bf16 NDHWC dense grid (coords are unique)
__global__ void scatter_feat(const float* __restrict__ vf, const int* __restrict__ coors,
                             unsigned short* __restrict__ act, int npts, int Ci,
                             int D, int H, int W) {
  int chunks = Ci >> 3;
  long long total = (long long)npts * chunks;
  for (long long i = blockIdx.x * (long long)blockDim.x + threadIdx.x; i < total;
       i += (long long)gridDim.x * blockDim.x) {
    int p = (int)(i / chunks);
    int c8 = (int)(i % chunks) << 3;
    int b = coors[4 * p + 0], d = coors[4 * p + 1], h = coors[4 * p + 2], w = coors[4 * p + 3];
    long long v = (((long long)(b * D + d) * H + h) * W + w);
    const float* src = vf + (long long)p * Ci + c8;
    unsigned short tmp[8];
#pragma unroll
    for (int j = 0; j < 8; ++j) tmp[j] = f2bf(src[j]);
    uint4 q;
    __builtin_memcpy(&q, tmp, 16);
    *(uint4*)(act + v * Ci + c8) = q;
  }
}
__global__ void scatter_mask(const int* __restrict__ coors, float* __restrict__ mask,
                             int npts, int D, int H, int W) {
  for (long long i = blockIdx.x * (long long)blockDim.x + threadIdx.x; i < npts;
       i += (long long)gridDim.x * blockDim.x) {
    int b = coors[4 * i + 0], d = coors[4 * i + 1], h = coors[4 * i + 2], w = coors[4 * i + 3];
    mask[(((long long)(b * D + d) * H + h) * W + w)] = 1.0f;
  }
}

// occupancy max-pool for strided layers
__global__ void down_mask_k(const float* __restrict__ mi, float* __restrict__ mo,
                            int B, int Di, int Hi, int Wi, int Do, int Ho, int Wo,
                            int sd, int sh, int sw) {
  long long total = (long long)B * Do * Ho * Wo;
  for (long long i = blockIdx.x * (long long)blockDim.x + threadIdx.x; i < total;
       i += (long long)gridDim.x * blockDim.x) {
    int x = (int)(i % Wo); long long r = i / Wo;
    int y = (int)(r % Ho); r /= Ho;
    int z = (int)(r % Do); int b = (int)(r / Do);
    float m = 0.0f;
    for (int dz = 0; dz < sd; ++dz)
      for (int dy = 0; dy < sh; ++dy)
        for (int dx = 0; dx < sw; ++dx) {
          float v = mi[(((long long)(b * Di + z * sd + dz) * Hi + y * sh + dy) * Wi + x * sw + dx)];
          m = fmaxf(m, v);
        }
    mo[i] = (m > 0.0f) ? 1.0f : 0.0f;
  }
}

// pack fp32 weights (O,I,kd,kh,kw) into bf16 B-fragment order:
// wp[(((t*nkc+kc)*NT + nt)*32 + lane)*16 + e], K = (e<8 ? (lane/16)*8 + e : 16 + (lane/16)*8 + e-8), N = lane%16
__global__ void pack_w(const float* __restrict__ W, unsigned short* __restrict__ wp,
                       int Cin, int Cout, int taps, int nkc, int NT) {
  long long total = (long long)taps * nkc * NT * 512;
  for (long long i = blockIdx.x * (long long)blockDim.x + threadIdx.x; i < total;
       i += (long long)gridDim.x * blockDim.x) {
    int e = (int)(i & 15);
    int lane = (int)((i >> 4) & 31);
    long long r = i >> 9;
    int nt = (int)(r % NT); r /= NT;
    int kc = (int)(r % nkc);
    int t = (int)(r / nkc);
    int g = lane >> 4, n = lane & 15;
    int K = (e < 8) ? (g * 8 + e) : (16 + g * 8 + (e - 8));
    int cin = kc * 32 + K;
    int cout = nt * 16 + n;
    float val = (cin < Cin) ? W[((long long)cout * Cin + cin) * taps + t] : 0.0f;
    wp[i] = f2bf(val);
  }
}

// ---------- implicit-GEMM conv via WMMA bf16 ----------
// one wave per tile of 16 consecutive output-W voxels; NT Cout-tiles share the A fragment
template <int NT>
__global__ void conv_wmma(const unsigned short* __restrict__ src,
                          unsigned short* __restrict__ dst,
                          const unsigned short* __restrict__ wp,
                          const float* __restrict__ mask_out,
                          int B, int Di, int Hi, int Wi, int Ci,
                          int Do, int Ho, int Wo, int Co,
                          int sd, int sh, int sw,
                          int KD, int KH, int KW,
                          int pd, int ph, int pw, int nkc) {
  int wave = (int)((blockIdx.x * (long long)blockDim.x + threadIdx.x) >> 5);
  int lane = threadIdx.x & 31;
  int wtiles = Wo >> 4;
  int mtiles = B * Do * Ho * wtiles;
  if (wave >= mtiles) return;
  int xt = wave % wtiles; int r1 = wave / wtiles;
  int yo = r1 % Ho; int r2 = r1 / Ho;
  int zo = r2 % Do; int b = r2 / Do;
  int row = lane & 15, g = lane >> 4;

  long long vbase = (((long long)(b * Do + zo) * Ho + yo) * Wo + ((long long)xt << 4));
  float mrow = mask_out[vbase + row];
  if (__ballot(mrow > 0.0f) == 0ull) return;   // whole output tile unoccupied

  v8f acc[NT] = {};
  int taps = KD * KH * KW;

  for (int t = 0; t < taps; ++t) {
    int kw_ = t % KW; int tt = t / KW;
    int kh_ = tt % KH; int kd_ = tt / KH;
    int zi = zo * sd + kd_ - pd;
    int yi = yo * sh + kh_ - ph;
    if ((unsigned)zi >= (unsigned)Di || (unsigned)yi >= (unsigned)Hi) continue; // uniform
    int xi = ((xt << 4) + row) * sw + kw_ - pw;
    bool vx = (unsigned)xi < (unsigned)Wi;
    const unsigned short* base =
        src + ((((long long)(b * Di + zi) * Hi + yi) * Wi + (vx ? xi : 0)) * Ci);
    const unsigned short* wbase = wp + ((long long)t * nkc) * (NT * 512) + lane * 16;
    for (int kc = 0; kc < nkc; ++kc) {
      int c0 = kc * 32 + g * 8;
      int c1 = c0 + 16;
      union { uint4 q[2]; v16bf v; } Au;
      uint4 zq = make_uint4(0u, 0u, 0u, 0u);
      Au.q[0] = (vx && c0 < Ci) ? *(const uint4*)(base + c0) : zq;
      Au.q[1] = (vx && c1 < Ci) ? *(const uint4*)(base + c1) : zq;
      const unsigned short* wk = wbase + (long long)kc * (NT * 512);
      __builtin_prefetch(wk + NT * 512, 0, 1);
#pragma unroll
      for (int nt = 0; nt < NT; ++nt) {
        union { uint4 q[2]; v16bf v; } Bu;
        const uint4* wq = (const uint4*)(wk + nt * 512);
        Bu.q[0] = wq[0];
        Bu.q[1] = wq[1];
        acc[nt] = __builtin_amdgcn_wmma_f32_16x16x32_bf16(
            false, Au.v, false, Bu.v, (short)0, acc[nt], false, false);
      }
    }
  }

  // C/D layout: VGPR r holds M = r + 8*g, column N = lane%16
#pragma unroll
  for (int nt = 0; nt < NT; ++nt) {
    int n = nt * 16 + row;
#pragma unroll
    for (int r = 0; r < 8; ++r) {
      int M = r + g * 8;
      dst[(vbase + M) * Co + n] = f2bf(acc[nt][r]);
    }
  }
}

// ---------- masked BN stats: st[0..C-1]=sum(m*y), st[C..2C-1]=sum(m*y^2), st[2C]=sum(m) ----------
__global__ void bn_stats(const unsigned short* __restrict__ y, const float* __restrict__ mask,
                         float* __restrict__ st, long long V, int C) {
  __shared__ float sh[129];
  int tid = threadIdx.x;
  int spc = 256 / C;                 // C in {16,32,64} divides 256
  int c = tid % C, slot = tid / C;
  float s1 = 0.f, s2 = 0.f, sm = 0.f;
  for (long long v = (long long)blockIdx.x * spc + slot; v < V;
       v += (long long)gridDim.x * spc) {
    float m = mask[v];
    float yv = bf2f(y[v * C + c]);
    float my = m * yv;
    s1 += my; s2 += my * yv;
    if (c == 0) sm += m;
  }
  if (tid < 2 * C + 1) sh[tid] = 0.0f;
  __syncthreads();
  atomicAdd(&sh[c], s1);
  atomicAdd(&sh[C + c], s2);
  if (c == 0) atomicAdd(&sh[2 * C], sm);
  __syncthreads();
  if (tid < 2 * C + 1) atomicAdd(&st[tid], sh[tid]);
}

// ---------- normalize + affine + ReLU + mask; optional final fp32 NCDHW write ----------
__global__ void bn_norm(const unsigned short* __restrict__ y, const float* __restrict__ mask,
                        const float* __restrict__ st, const float* __restrict__ gamma,
                        const float* __restrict__ beta, unsigned short* __restrict__ actout,
                        float* __restrict__ finalout, long long V, int C, long long DHW) {
  long long total = V * C;
  for (long long i = blockIdx.x * (long long)blockDim.x + threadIdx.x; i < total;
       i += (long long)gridDim.x * blockDim.x) {
    long long v = i / C;
    int c = (int)(i % C);
    float cnt = st[2 * C];
    float mean = st[c] / cnt;
    float var = st[C + c] / cnt - mean * mean;
    float m = mask[v];
    float out = (bf2f(y[i]) - mean) * rsqrtf(var + 1e-5f) * gamma[c] + beta[c];
    out = fmaxf(out, 0.0f) * m;
    if (actout) actout[i] = f2bf(out);
    if (finalout) {
      long long b = v / DHW, sp = v % DHW;
      finalout[((long long)b * C + c) * DHW + sp] = out;
    }
  }
}

// ---------- host driver ----------
struct LayerCfg { int Ci, Co, kd, kh, kw, sd, sh, sw; };

extern "C" void kernel_launch(void* const* d_in, const int* in_sizes, int n_in,
                              void* d_out, int out_size, void* d_ws, size_t ws_size,
                              hipStream_t stream) {
  (void)n_in; (void)out_size; (void)ws_size;
  static const LayerCfg Ls[9] = {
      {128, 16, 3, 3, 3, 1, 1, 1}, {16, 16, 3, 3, 3, 1, 1, 1},
      {16, 32, 3, 3, 3, 2, 2, 2},  {32, 32, 3, 3, 3, 1, 1, 1},
      {32, 64, 3, 3, 3, 2, 2, 2},  {64, 64, 3, 3, 3, 1, 1, 1},
      {64, 64, 3, 3, 3, 2, 2, 2},  {64, 64, 3, 3, 3, 1, 1, 1},
      {64, 64, 3, 1, 1, 2, 1, 1}};
  const int Bb = 2;
  const float* vf = (const float*)d_in[0];
  const int* coors = (const int*)d_in[1];
  const int npts = in_sizes[0] / 128;

  // workspace carve-up
  char* ws = (char*)d_ws;
  size_t off = 0;
  auto take = [&](size_t bytes) { size_t o = off; off = (off + bytes + 255) & ~(size_t)255; return o; };
  unsigned short* bufA = (unsigned short*)(ws + take(524288ull * 128 * 2)); // 134 MB bf16
  unsigned short* bufB = (unsigned short*)(ws + take(524288ull * 16 * 2));  // 17 MB bf16
  float* m[5];
  const long long mV[5] = {2LL * 16 * 128 * 128, 2LL * 8 * 64 * 64, 2LL * 4 * 32 * 32,
                           2LL * 2 * 16 * 16, 2LL * 1 * 16 * 16};
  for (int i = 0; i < 5; ++i) m[i] = (float*)(ws + take((size_t)mV[i] * 4));
  unsigned short* wp = (unsigned short*)(ws + take(262144));
  float* st = (float*)(ws + take(1024));

  // densify: zero input grid + mask, scatter features and occupancy
  zero_u4<<<4096, 256, 0, stream>>>((uint4*)bufA, 524288LL * 128 / 8);
  zero_f32<<<512, 256, 0, stream>>>(m[0], mV[0]);
  {
    long long sc = (long long)npts * (128 / 8);
    scatter_feat<<<(unsigned)((sc + 255) / 256), 256, 0, stream>>>(vf, coors, bufA, npts, 128, 16, 128, 128);
    scatter_mask<<<(unsigned)((npts + 255) / 256), 256, 0, stream>>>(coors, m[0], npts, 16, 128, 128);
  }

  int Di = 16, Hi = 128, Wi = 128;
  int lvl = 0;
  unsigned short* src = bufA;
  unsigned short* dst = bufB;
  for (int i = 0; i < 9; ++i) {
    const LayerCfg& L = Ls[i];
    int Do = Di / L.sd, Ho = Hi / L.sh, Wo = Wi / L.sw;
    int taps = L.kd * L.kh * L.kw;
    int nkc = (L.Ci + 31) / 32;
    int NT = L.Co / 16;
    int pd = L.kd / 2, ph = L.kh / 2, pw = L.kw / 2;
    long long V = (long long)Bb * Do * Ho * Wo;
    const float* Wf = (const float*)d_in[4 + i];
    const float* gam = (const float*)d_in[13 + i];
    const float* bet = (const float*)d_in[22 + i];

    long long ptotal = (long long)taps * nkc * NT * 512;
    pack_w<<<(unsigned)((ptotal + 255) / 256), 256, 0, stream>>>(Wf, wp, L.Ci, L.Co, taps, nkc, NT);

    if (L.sd != 1 || L.sh != 1 || L.sw != 1) {
      ++lvl;
      down_mask_k<<<(unsigned)((V + 255) / 256), 256, 0, stream>>>(
          m[lvl - 1], m[lvl], Bb, Di, Hi, Wi, Do, Ho, Wo, L.sd, L.sh, L.sw);
    }
    const float* mOut = m[lvl];

    int mtiles = Bb * Do * Ho * (Wo >> 4);
    unsigned blocks = (unsigned)((mtiles + 3) / 4);  // 4 waves / 128-thread block
    switch (NT) {
      case 1:
        conv_wmma<1><<<blocks, 128, 0, stream>>>(src, dst, wp, mOut, Bb, Di, Hi, Wi, L.Ci,
                                                 Do, Ho, Wo, L.Co, L.sd, L.sh, L.sw,
                                                 L.kd, L.kh, L.kw, pd, ph, pw, nkc);
        break;
      case 2:
        conv_wmma<2><<<blocks, 128, 0, stream>>>(src, dst, wp, mOut, Bb, Di, Hi, Wi, L.Ci,
                                                 Do, Ho, Wo, L.Co, L.sd, L.sh, L.sw,
                                                 L.kd, L.kh, L.kw, pd, ph, pw, nkc);
        break;
      default:
        conv_wmma<4><<<blocks, 128, 0, stream>>>(src, dst, wp, mOut, Bb, Di, Hi, Wi, L.Ci,
                                                 Do, Ho, Wo, L.Co, L.sd, L.sh, L.sw,
                                                 L.kd, L.kh, L.kw, pd, ph, pw, nkc);
        break;
    }

    zero_f32<<<1, 256, 0, stream>>>(st, 2 * L.Co + 1);
    bn_stats<<<512, 256, 0, stream>>>(dst, mOut, st, V, L.Co);

    long long nTot = V * L.Co;
    unsigned nBlocks = (unsigned)((nTot + 255) / 256);
    if (i == 8) {
      bn_norm<<<nBlocks, 256, 0, stream>>>(dst, mOut, st, gam, bet, (unsigned short*)nullptr,
                                           (float*)d_out, V, L.Co, (long long)Do * Ho * Wo);
    } else {
      bn_norm<<<nBlocks, 256, 0, stream>>>(dst, mOut, st, gam, bet, dst, (float*)nullptr,
                                           V, L.Co, (long long)Do * Ho * Wo);
    }

    // ping-pong
    unsigned short* nsrc = dst;
    dst = (dst == bufB) ? bufA : bufB;
    src = nsrc;
    Di = Do; Hi = Ho; Wi = Wo;
  }
}
